// BlockwiseAttention_61117384622785
// MI455X (gfx1250) — matrix operand order repaired and checked by reading, not compiled, not run on hardware
//
#include <hip/hip_runtime.h>
#include <hip/hip_bf16.h>

// Problem constants (from reference setup_inputs)
#define B_   4
#define H_   16
#define T_   4096
#define D_   128
#define BS_  32
#define C_   (T_ / BS_)          // 128
#define LDSS 136                 // LDS row stride in bf16 elems (16B-aligned rows, padded)

typedef float  v8f   __attribute__((ext_vector_type(8)));
typedef __bf16 v16bf __attribute__((ext_vector_type(16)));
typedef int    v4i   __attribute__((ext_vector_type(4)));

// gfx1250 async global->LDS copy path (ASYNCcnt), if this toolchain declares it.
#if defined(__HIP_DEVICE_COMPILE__) &&                                   \
    __has_builtin(__builtin_amdgcn_global_load_async_to_lds_b128) &&     \
    __has_builtin(__builtin_amdgcn_s_wait_asynccnt)
#define USE_ASYNC_LDS 1
#else
#define USE_ASYNC_LDS 0
#endif

// Builtin signature (from hipcc diagnostics):
//   (v4i* src_global_generic, __attribute__((address_space(3))) v4i* dst_lds,
//    const int imm_offset, const int imm_cpol)
#define GSRC(p) ((v4i*)(p))
#define AS3V(p) ((__attribute__((address_space(3))) v4i*)(p))

// ---------------------------------------------------------------------------
// Mean-pool compression: [BH, T, D] -> bf16 [BH, C, D] (or transposed
// [BH, D, C] when xpose != 0, used for V so the attention kernel can stage it
// into LDS with straight row copies / async DMA).
// ---------------------------------------------------------------------------
__global__ void nsa_compress_kernel(const float* __restrict__ src,
                                    __bf16* __restrict__ dst, int n, int xpose) {
    int idx = blockIdx.x * blockDim.x + threadIdx.x;
    if (idx >= n) return;
    int d   = idx & (D_ - 1);
    int bhc = idx >> 7;                       // (bh*C + c)
    int c   = bhc & (C_ - 1);
    int bh  = idx >> 14;
    const float* p = src + (size_t)bhc * BS_ * D_ + d;
    float s = 0.0f;
#pragma unroll
    for (int i = 0; i < BS_; ++i) s += p[i * D_];
    __bf16 r = (__bf16)(s * (1.0f / BS_));
    size_t o = xpose ? (((size_t)bh * D_ + d) * C_ + c) : (size_t)idx;
    dst[o] = r;
}

// ---------------------------------------------------------------------------
// Fragment loaders (ISA 7.12.2, 16-bit data, wave32):
//   A 16x32: lane m holds row m; lanes 0-15 K = {k0..k0+7, k0+16..k0+23},
//            lanes 16-31 K = {k0+8..k0+15, k0+24..k0+31}.
//   B 32x16: lane n holds column n; same K split as A:
//            lanes 0-15 K = {k0..k0+7, k0+16..k0+23}, lanes 16-31 K = {k0+8..k0+15, k0+24..k0+31}.
// ---------------------------------------------------------------------------
__device__ __forceinline__ v16bf ld_fragA(const __bf16* sm, int row0, int k0, int lane) {
    const __bf16* p = sm + (row0 + (lane & 15)) * LDSS + k0 + ((lane >> 4) << 3);
    union { uint4 u[2]; v16bf v; } t;
    t.u[0] = *(const uint4*)(p);
    t.u[1] = *(const uint4*)(p + 16);
    return t.v;
}

__device__ __forceinline__ v16bf ld_fragB(const __bf16* sm, int col0, int k0, int lane) {
    const __bf16* p = sm + (col0 + (lane & 15)) * LDSS + k0 + ((lane >> 4) << 3);
    union { uint4 u[2]; v16bf v; } t;
    t.u[0] = *(const uint4*)(p);
    t.u[1] = *(const uint4*)(p + 16);
    return t.v;
}

// ---------------------------------------------------------------------------
// Attention over compressed KV. One block = 8 waves = 128 q rows of one (b,h).
//   grid = (T/128, B*H), block = 256, dynamic LDS = 4 * 128 * LDSS * 2 bytes.
// ---------------------------------------------------------------------------
__global__ __launch_bounds__(256)
void nsa_attn_kernel(const float*  __restrict__ q,
                     const __bf16* __restrict__ kcmp,
                     const __bf16* __restrict__ vcmpT,
                     float* __restrict__ out,
                     float* __restrict__ score) {
    extern __shared__ __align__(16) char smem[];
    __bf16* sQ  = (__bf16*)smem;                 // 128 x LDSS  (q tile, bf16)
    __bf16* sK  = sQ  + 128 * LDSS;              // 128 x LDSS  (k_cmp row-major [c][d])
    __bf16* sVT = sK  + 128 * LDSS;              // 128 x LDSS  (v_cmp transposed [d][c])
    __bf16* sP  = sVT + 128 * LDSS;              // 128 x LDSS  (softmax probs, bf16)

    const int tid  = threadIdx.x;
    const int lane = tid & 31;
    const int wave = tid >> 5;
    const int tile = blockIdx.x;                 // q-row tile (128 rows)
    const int bh   = blockIdx.y;                 // b*H + h

    const size_t qbase = ((size_t)bh * T_ + (size_t)tile * 128) * D_;
    const size_t cbase = (size_t)bh * C_ * D_;   // base for both kcmp and vcmpT

    __builtin_prefetch(q + qbase, 0, 1);         // global_prefetch_b8

    // ---- Stage K / V^T tiles into LDS (async DMA when available) ----------
#if USE_ASYNC_LDS
#pragma unroll
    for (int i = 0; i < 8; ++i) {                // 2048 chunks of 16B per tile
        int e = tid + i * 256;
        int r = e >> 4;                          // row (c for K, d for V^T)
        int c = (e & 15) << 3;                   // col start, 8 bf16 = 16B
        __builtin_amdgcn_global_load_async_to_lds_b128(
            GSRC(kcmp + cbase + r * D_ + c),  AS3V(sK + r * LDSS + c), 0, 0);
        __builtin_amdgcn_global_load_async_to_lds_b128(
            GSRC(vcmpT + cbase + r * C_ + c), AS3V(sVT + r * LDSS + c), 0, 0);
    }
#else
#pragma unroll
    for (int i = 0; i < 64; ++i) {
        int e = tid + i * 256;
        int r = e >> 7, c = e & 127;
        sK[r * LDSS + c]  = kcmp[cbase + e];
        sVT[r * LDSS + c] = vcmpT[cbase + e];
    }
#endif

    // ---- Stage Q tile (fp32 -> bf16 conversion, overlaps async DMA) -------
#pragma unroll
    for (int i = 0; i < 64; ++i) {               // 128*128 elems / 256 threads
        int e = tid + i * 256;
        int r = e >> 7, c = e & 127;
        sQ[r * LDSS + c] = (__bf16)q[qbase + e];
    }

#if USE_ASYNC_LDS
    __builtin_amdgcn_s_wait_asynccnt(0);
#endif
    __syncthreads();

    const int row0 = wave * 16;                  // this wave's 16 q rows (in tile)

    // ---- S = Q * K_cmp^T  (8 tiles of 16x16, K over D=128 in 4 chunks) ----
    v16bf aq[4];
#pragma unroll
    for (int kk = 0; kk < 4; ++kk) aq[kk] = ld_fragA(sQ, row0, kk * 32, lane);

    v8f accS[8];
#pragma unroll
    for (int t = 0; t < 8; ++t) {
        v8f acc = {0.f, 0.f, 0.f, 0.f, 0.f, 0.f, 0.f, 0.f};
#pragma unroll
        for (int kk = 0; kk < 4; ++kk) {
            v16bf b = ld_fragB(sK, t * 16, kk * 32, lane);
            acc = __builtin_amdgcn_wmma_f32_16x16x32_bf16(
                      false, aq[kk], false, b, (short)0, acc, false, false);
        }
        accS[t] = acc;
    }

    // ---- Row softmax in registers + emit score (fp32) and P (bf16) --------
    // C/D layout: element [M][N] with M = r + 8*(lane>=16), N = lane&15.
    const float scale = 0.08838834764831845f;    // 1/sqrt(128)
    const int ncol = lane & 15;
    const int mhi  = (lane >> 4) << 3;
    const size_t srow_base = ((size_t)bh * T_ + (size_t)tile * 128 + row0) * (size_t)C_;

#pragma unroll
    for (int r = 0; r < 8; ++r) {
        float v[8];
        float m = -3.0e38f;
#pragma unroll
        for (int t = 0; t < 8; ++t) { v[t] = accS[t][r] * scale; m = fmaxf(m, v[t]); }
#pragma unroll
        for (int msk = 8; msk >= 1; msk >>= 1) m = fmaxf(m, __shfl_xor(m, msk, 32));
        float s = 0.0f;
#pragma unroll
        for (int t = 0; t < 8; ++t) { v[t] = __expf(v[t] - m); s += v[t]; }
#pragma unroll
        for (int msk = 8; msk >= 1; msk >>= 1) s += __shfl_xor(s, msk, 32);
        const float inv  = 1.0f / s;
        const int   mrow = r + mhi;
        const size_t so  = srow_base + (size_t)mrow * C_ + ncol;
#pragma unroll
        for (int t = 0; t < 8; ++t) {
            float p = v[t] * inv;
            score[so + t * 16] = p;
            sP[(row0 + mrow) * LDSS + t * 16 + ncol] = (__bf16)p;
        }
    }
    __syncthreads();

    // ---- O = P * V_cmp  (B-fragments come from transposed V in LDS) -------
    v16bf ap[4];
#pragma unroll
    for (int kk = 0; kk < 4; ++kk) ap[kk] = ld_fragA(sP, row0, kk * 32, lane);

    const size_t orow_base = ((size_t)bh * T_ + (size_t)tile * 128 + row0) * (size_t)D_;
#pragma unroll
    for (int dt = 0; dt < 8; ++dt) {
        v8f acc = {0.f, 0.f, 0.f, 0.f, 0.f, 0.f, 0.f, 0.f};
#pragma unroll
        for (int kk = 0; kk < 4; ++kk) {
            v16bf b = ld_fragB(sVT, dt * 16, kk * 32, lane);
            acc = __builtin_amdgcn_wmma_f32_16x16x32_bf16(
                      false, ap[kk], false, b, (short)0, acc, false, false);
        }
#pragma unroll
        for (int r = 0; r < 8; ++r)
            out[orow_base + (size_t)(r + mhi) * D_ + dt * 16 + ncol] = acc[r];
    }
}

// ---------------------------------------------------------------------------
extern "C" void kernel_launch(void* const* d_in, const int* in_sizes, int n_in,
                              void* d_out, int out_size, void* d_ws, size_t ws_size,
                              hipStream_t stream) {
    const float* q = (const float*)d_in[0];
    const float* k = (const float*)d_in[1];
    const float* v = (const float*)d_in[2];

    float* out   = (float*)d_out;
    float* score = out + (size_t)B_ * H_ * T_ * D_;

    __bf16* kcmp  = (__bf16*)d_ws;                      // BH*C*D bf16 = 2 MB
    __bf16* vcmpT = kcmp + (size_t)B_ * H_ * C_ * D_;   // +2 MB (stored [bh][d][c])

    const int nComp = B_ * H_ * C_ * D_;                // 1,048,576
    nsa_compress_kernel<<<(nComp + 255) / 256, 256, 0, stream>>>(k, kcmp,  nComp, 0);
    nsa_compress_kernel<<<(nComp + 255) / 256, 256, 0, stream>>>(v, vcmpT, nComp, 1);

    dim3 grid(T_ / 128, B_ * H_);
    const size_t lds_bytes = (size_t)4 * 128 * LDSS * sizeof(__bf16);  // ~136 KB < 320 KB
    nsa_attn_kernel<<<grid, 256, lds_bytes, stream>>>(q, kcmp, vcmpT, out, score);
}
